// CubeGatedBlock_65755949301988
// MI455X (gfx1250) — matrix operand order repaired and checked by reading, not compiled, not run on hardware
//
#include <hip/hip_runtime.h>
#include <hip/hip_bf16.h>
#include <math.h>

#define BL      8192
#define D       1024
#define NSLOTS  32768
#define TOPK    16
#define NBLK    (BL / 16)     // 512 row-blocks of 16 (proj/gate)
#define NBLK32  (BL / 32)     // 256 row-blocks of 32 (sims)

typedef __attribute__((ext_vector_type(16))) __bf16       v16bf;
typedef __attribute__((ext_vector_type(8)))  float        v8f;
typedef __attribute__((ext_vector_type(8)))  unsigned int v8u;

__device__ __forceinline__ unsigned short f2bf(float f) {
    unsigned int u = __builtin_bit_cast(unsigned int, f);
    u += 0x7fffu + ((u >> 16) & 1u);          // round-to-nearest-even
    return (unsigned short)(u >> 16);
}

// A fragment (16-bit A, 16x32). Per ISA 7.12.2: lane m=L%16;
// half 0: v0..3 = K 0..7, v4..7 = K 16..23; half 1: K 8..15 / 24..31.
__device__ __forceinline__ v16bf frag_a(const unsigned int* row, int kt, int hf) {
    uint4 lo = *(const uint4*)(row + kt * 16 + hf * 4);
    uint4 hi = *(const uint4*)(row + kt * 16 + 8 + hf * 4);
    v8u u;
    u[0] = lo.x; u[1] = lo.y; u[2] = lo.z; u[3] = lo.w;
    u[4] = hi.x; u[5] = hi.y; u[6] = hi.z; u[7] = hi.w;
    return __builtin_bit_cast(v16bf, u);
}

// B fragment (16-bit B, 32x16). Source is BT row-major: B(k,n) = BT[n][k].
// lane n=L%16; half 0: K 0..7 / 16..23; half 1: K 8..15 / 24..31 (matches A).
__device__ __forceinline__ v16bf frag_b(const unsigned int* row, int kt, int hf) {
    uint4 lo = *(const uint4*)(row + kt * 16 + hf * 4);
    uint4 hi = *(const uint4*)(row + kt * 16 + 8 + hf * 4);
    v8u u;
    u[0] = lo.x; u[1] = lo.y; u[2] = lo.z; u[3] = lo.w;
    u[4] = hi.x; u[5] = hi.y; u[6] = hi.z; u[7] = hi.w;
    return __builtin_bit_cast(v16bf, u);
}

#define WMMA_BF16(a, b, c) \
    __builtin_amdgcn_wmma_f32_16x16x32_bf16(false, (a), false, (b), (short)0, (c), false, false)

// 4-way N-blocked MMA: one shared A fragment per k-step feeds 4 B tiles.
template <int NKT>
__device__ __forceinline__ void mma_n4(const unsigned int* arow,
                                       const unsigned int* b0, const unsigned int* b1,
                                       const unsigned int* b2, const unsigned int* b3,
                                       int hf, v8f& c0, v8f& c1, v8f& c2, v8f& c3) {
    for (int kt = 0; kt < NKT; ++kt) {
        v16bf a = frag_a(arow, kt, hf);
        c0 = WMMA_BF16(a, frag_b(b0, kt, hf), c0);
        c1 = WMMA_BF16(a, frag_b(b1, kt, hf), c1);
        c2 = WMMA_BF16(a, frag_b(b2, kt, hf), c2);
        c3 = WMMA_BF16(a, frag_b(b3, kt, hf), c3);
    }
}

// 2x4 M,N-blocked MMA: 2 A frags + 4 B frags -> 8 WMMAs per k-step
// (1.5 b128 loads per WMMA).
template <int NKT>
__device__ __forceinline__ void mma_m2n4(const unsigned int* a0r, const unsigned int* a1r,
                                         const unsigned int* b0, const unsigned int* b1,
                                         const unsigned int* b2, const unsigned int* b3,
                                         int hf, v8f c[2][4]) {
    for (int kt = 0; kt < NKT; ++kt) {
        v16bf a0 = frag_a(a0r, kt, hf);
        v16bf a1 = frag_a(a1r, kt, hf);
        v16bf f0 = frag_b(b0, kt, hf);
        v16bf f1 = frag_b(b1, kt, hf);
        v16bf f2 = frag_b(b2, kt, hf);
        v16bf f3 = frag_b(b3, kt, hf);
        c[0][0] = WMMA_BF16(a0, f0, c[0][0]);
        c[1][0] = WMMA_BF16(a1, f0, c[1][0]);
        c[0][1] = WMMA_BF16(a0, f1, c[0][1]);
        c[1][1] = WMMA_BF16(a1, f1, c[1][1]);
        c[0][2] = WMMA_BF16(a0, f2, c[0][2]);
        c[1][2] = WMMA_BF16(a1, f2, c[1][2]);
        c[0][3] = WMMA_BF16(a0, f3, c[0][3]);
        c[1][3] = WMMA_BF16(a1, f3, c[1][3]);
    }
}

// ---------------- K0: zero the two scalar outputs (atomic accumulators) -------
__global__ void cgb_init(float* tail) {
    if (threadIdx.x < 2) tail[threadIdx.x] = 0.f;
}

// ---------------- K1: h_in f32 -> bf16 (packed pairs) -------------------------
__global__ void cgb_cvt_h(const float* __restrict__ h, unsigned int* __restrict__ hb) {
    int i = blockIdx.x * blockDim.x + threadIdx.x;    // over BL*D/2
    float a = h[2 * i], b = h[2 * i + 1];
    hb[i] = (unsigned int)f2bf(a) | ((unsigned int)f2bf(b) << 16);
}

// ---------------- K2: WkT bf16: wkT[n][k] = Wk[k][n] --------------------------
__global__ void cgb_cvt_wkT(const float* __restrict__ Wk, unsigned int* __restrict__ wkT) {
    int i = blockIdx.x * blockDim.x + threadIdx.x;    // over D*(D/2)
    int n = i / (D / 2), kk = i % (D / 2), k0 = 2 * kk;
    wkT[(size_t)n * (D / 2) + kk] =
        (unsigned int)f2bf(Wk[(size_t)k0 * D + n]) |
        ((unsigned int)f2bf(Wk[(size_t)(k0 + 1) * D + n]) << 16);
}

// ---------------- K3: W1T bf16 (first 2048 rows): w1T[n][k] = W1[k][n] --------
__global__ void cgb_cvt_w1T(const float* __restrict__ W1, unsigned int* __restrict__ w1T) {
    int i = blockIdx.x * blockDim.x + threadIdx.x;    // over D*1024
    int n = i / 1024, kk = i % 1024, k0 = 2 * kk;
    w1T[(size_t)n * 1024 + kk] =
        (unsigned int)f2bf(W1[(size_t)k0 * D + n]) |
        ((unsigned int)f2bf(W1[(size_t)(k0 + 1) * D + n]) << 16);
}

// ---------------- K4: row-normalize cube_keys -> mn bf16 ----------------------
__global__ __launch_bounds__(256) void cgb_norm_cube(const float* __restrict__ ck,
                                                     unsigned int* __restrict__ mn) {
    int row = blockIdx.x, t = threadIdx.x;
    const float* r = ck + (size_t)row * D;
    float4 v = ((const float4*)r)[t];
    __shared__ float red[256];
    red[t] = v.x * v.x + v.y * v.y + v.z * v.z + v.w * v.w;
    __syncthreads();
    for (int s = 128; s > 0; s >>= 1) {
        if (t < s) red[t] += red[t + s];
        __syncthreads();
    }
    float inv = rsqrtf(red[0] + 1e-8f);
    uint2 o;
    o.x = (unsigned int)f2bf(v.x * inv) | ((unsigned int)f2bf(v.y * inv) << 16);
    o.y = (unsigned int)f2bf(v.z * inv) | ((unsigned int)f2bf(v.w * inv) << 16);
    ((uint2*)(mn + (size_t)row * (D / 2)))[t] = o;
}

// ---------------- K5: keys = h @ Wk + bk, row-normalize -> kn bf16 ------------
__global__ __launch_bounds__(128) void cgb_proj(const unsigned int* __restrict__ hb,
                                                const unsigned int* __restrict__ wkT,
                                                const float* __restrict__ bk,
                                                unsigned int* __restrict__ kn) {
    __shared__ float keys[16][D];
    __shared__ float rred[16][8];
    __shared__ float rnorm[16];
    int tid = threadIdx.x, wv = tid >> 5, lane = tid & 31;
    int ln = lane & 15, hf = lane >> 4;
    int m0 = blockIdx.x * 16;
    const unsigned int* arow = hb + (size_t)(m0 + ln) * (D / 2);
    for (int g = 0; g < 4; ++g) {
        int nt0 = wv * 16 + g * 4;
        const unsigned int* b0 = wkT + (size_t)((nt0 + 0) * 16 + ln) * (D / 2);
        const unsigned int* b1 = wkT + (size_t)((nt0 + 1) * 16 + ln) * (D / 2);
        const unsigned int* b2 = wkT + (size_t)((nt0 + 2) * 16 + ln) * (D / 2);
        const unsigned int* b3 = wkT + (size_t)((nt0 + 3) * 16 + ln) * (D / 2);
        v8f c0 = {}, c1 = {}, c2 = {}, c3 = {};
        mma_n4<32>(arow, b0, b1, b2, b3, hf, c0, c1, c2, c3);
#pragma unroll
        for (int i = 0; i < 4; ++i) {
            int n = (nt0 + i) * 16 + ln;
            float bias = bk[n];
            v8f c = (i == 0) ? c0 : (i == 1) ? c1 : (i == 2) ? c2 : c3;
#pragma unroll
            for (int v = 0; v < 8; ++v) keys[hf * 8 + v][n] = c[v] + bias;
        }
    }
    __syncthreads();
    {   // row sum of squares: 8 threads per row
        int r = tid >> 3, p = tid & 7;
        float ss = 0.f;
        for (int c = p * 128; c < p * 128 + 128; ++c) { float x = keys[r][c]; ss += x * x; }
        rred[r][p] = ss;
    }
    __syncthreads();
    if (tid < 16) {
        float s = 0.f;
        for (int p = 0; p < 8; ++p) s += rred[tid][p];
        rnorm[tid] = rsqrtf(s + 1e-8f);
    }
    __syncthreads();
    {
        int r = tid >> 3, p = tid & 7;
        float inv = rnorm[r];
        unsigned int* orow = kn + (size_t)(m0 + r) * (D / 2) + p * 64;
        for (int c = 0; c < 64; ++c) {
            float x0 = keys[r][p * 128 + 2 * c] * inv;
            float x1 = keys[r][p * 128 + 2 * c + 1] * inv;
            orow[c] = (unsigned int)f2bf(x0) | ((unsigned int)f2bf(x1) << 16);
        }
    }
}

// ------- K6: sims GEMM (32 rows/block, 2x4 blocked) + fused top-16 +
//         softmax + gather -> pred, conf ---------------------------------------
__global__ __launch_bounds__(128) void cgb_sims_topk(const unsigned int* __restrict__ kn,
                                                     const unsigned int* __restrict__ mn,
                                                     const float* __restrict__ cvals,
                                                     float* __restrict__ pred,
                                                     float* __restrict__ confg,
                                                     float* __restrict__ mean_out) {
    __shared__ float tval[4][32][16];     //  8 KB
    __shared__ int   tidxs[4][32][16];    //  8 KB
    __shared__ float stage[4][32][64];    // 32 KB
    __shared__ float mval[32][16];
    __shared__ int   midx[32][16];
    __shared__ float wts[32][16];
    int tid = threadIdx.x, wv = tid >> 5, lane = tid & 31;
    int ln = lane & 15, hf = lane >> 4;
    int m0 = blockIdx.x * 32;
    for (int i = tid; i < 4 * 32 * 16; i += 128) {
        ((float*)tval)[i] = -1e30f; ((int*)tidxs)[i] = 0;
    }
    __syncthreads();
    const unsigned int* a0r = kn + (size_t)(m0 + ln) * (D / 2);
    const unsigned int* a1r = kn + (size_t)(m0 + 16 + ln) * (D / 2);
    int base0 = wv * (NSLOTS / 4);
    for (int t = 0; t < NSLOTS / 4 / 64; ++t) {   // 128 iterations of 64 slots
        int nb = base0 + t * 64;
        const unsigned int* b0 = mn + (size_t)(nb + 0 * 16 + ln) * (D / 2);
        const unsigned int* b1 = mn + (size_t)(nb + 1 * 16 + ln) * (D / 2);
        const unsigned int* b2 = mn + (size_t)(nb + 2 * 16 + ln) * (D / 2);
        const unsigned int* b3 = mn + (size_t)(nb + 3 * 16 + ln) * (D / 2);
        if (t + 1 < NSLOTS / 4 / 64) {   // pull next B rows toward the WGP
            const unsigned int* nx = mn + (size_t)(nb + 64 + lane) * (D / 2);
            __builtin_prefetch(nx, 0, 1);
            __builtin_prefetch(nx + 32 * (D / 2), 0, 1);
        }
        v8f c[2][4] = {};
        mma_m2n4<32>(a0r, a1r, b0, b1, b2, b3, hf, c);
#pragma unroll
        for (int mi = 0; mi < 2; ++mi)
#pragma unroll
            for (int ni = 0; ni < 4; ++ni)
#pragma unroll
                for (int v = 0; v < 8; ++v)
                    stage[wv][mi * 16 + hf * 8 + v][ni * 16 + ln] = c[mi][ni][v];
        // per-wave top-16 maintenance: all 32 lanes, one lane per row
        {
            int r = lane;
            float vmin = tval[wv][r][15];
            for (int cc = 0; cc < 64; ++cc) {
                float v = stage[wv][r][cc];
                if (v > vmin) {
                    int p = 15;
                    while (p > 0 && tval[wv][r][p - 1] < v) {
                        tval[wv][r][p] = tval[wv][r][p - 1];
                        tidxs[wv][r][p] = tidxs[wv][r][p - 1];
                        --p;
                    }
                    tval[wv][r][p] = v;
                    tidxs[wv][r][p] = nb + cc;
                    vmin = tval[wv][r][15];
                }
            }
        }
    }
    __syncthreads();
    if (tid < 32) {   // merge 4 partial lists (64 candidates) -> top-16, softmax
        int r = tid;
        for (int k = 0; k < TOPK; ++k) {
            float best = -1e30f; int bw = 0, bp = 0;
            for (int w = 0; w < 4; ++w)
                for (int p = 0; p < 16; ++p) {
                    float v = tval[w][r][p];
                    if (v > best) { best = v; bw = w; bp = p; }
                }
            mval[r][k] = best;
            midx[r][k] = tidxs[bw][r][bp];
            tval[bw][r][bp] = -1e30f;
        }
        float v0 = mval[r][0], s = 0.f;
        for (int k = 0; k < TOPK; ++k) {
            float e = expf((mval[r][k] - v0) * 10.0f);   // 1/TEMP = 10
            wts[r][k] = e; s += e;
        }
        float inv = 1.f / s;
        for (int k = 0; k < TOPK; ++k) wts[r][k] *= inv;
        float cf = (v0 + 1.f) * 0.5f;
        confg[m0 + r] = cf;
        atomicAdd(mean_out + 1, cf * (1.f / (float)BL));
    }
    __syncthreads();
    {   // gather + weighted sum: each thread handles 8 contiguous value dims
        int c0 = tid * 8;
        for (int r = 0; r < 32; ++r) {
            float acc[8];
#pragma unroll
            for (int i = 0; i < 8; ++i) acc[i] = 0.f;
            for (int k = 0; k < TOPK; ++k) {
                float w = wts[r][k];
                int id = midx[r][k];
                const float4* vp = (const float4*)(cvals + (size_t)id * D + c0);
                float4 a = vp[0], b = vp[1];
                acc[0] += w * a.x; acc[1] += w * a.y; acc[2] += w * a.z; acc[3] += w * a.w;
                acc[4] += w * b.x; acc[5] += w * b.y; acc[6] += w * b.z; acc[7] += w * b.w;
            }
            float* op = pred + (size_t)(m0 + r) * D + c0;
#pragma unroll
            for (int i = 0; i < 8; ++i) op[i] = acc[i];
        }
    }
}

// ------- K7: gate MLP (WMMA) + blend + mean(alpha) ----------------------------
__global__ __launch_bounds__(128) void cgb_gate(const float* __restrict__ h,
                                                const float* __restrict__ pred,
                                                const float* __restrict__ confg,
                                                const unsigned int* __restrict__ w1T,
                                                const float* __restrict__ W1f,
                                                const float* __restrict__ b1,
                                                const float* __restrict__ W2,
                                                const float* __restrict__ b2,
                                                const float* __restrict__ g_in,
                                                const float* __restrict__ bi_in,
                                                const float* __restrict__ g_pr,
                                                const float* __restrict__ bi_pr,
                                                float* __restrict__ yout,
                                                float* __restrict__ mean_out) {
    __shared__ __align__(16) unsigned short feats[16][2048];   // bf16 [LN(h) | LN(pred)]
    __shared__ float hmid[16][D];
    __shared__ float red[16][8][2];
    __shared__ float stats[16][2];
    __shared__ float confl[16];
    __shared__ float alphal[16];
    __shared__ float dotred[16][8];
    int tid = threadIdx.x, wv = tid >> 5, lane = tid & 31;
    int ln = lane & 15, hf = lane >> 4;
    int m0 = blockIdx.x * 16;

    // --- LayerNorm(h_in) -> feats[:, 0:1024] (bf16)
    {
        int r = tid >> 3, p = tid & 7;
        const float* row = h + (size_t)(m0 + r) * D;
        float s = 0.f, ss = 0.f;
        for (int c = p * 128; c < p * 128 + 128; ++c) { float x = row[c]; s += x; ss += x * x; }
        red[r][p][0] = s; red[r][p][1] = ss;
    }
    __syncthreads();
    if (tid < 16) {
        float s = 0.f, ss = 0.f;
        for (int p = 0; p < 8; ++p) { s += red[tid][p][0]; ss += red[tid][p][1]; }
        float mu = s * (1.f / D), var = ss * (1.f / D) - mu * mu;
        stats[tid][0] = mu; stats[tid][1] = rsqrtf(var + 1e-5f);
        confl[tid] = confg[m0 + tid];
    }
    __syncthreads();
    {
        int r = tid >> 3, p = tid & 7;
        const float* row = h + (size_t)(m0 + r) * D;
        float mu = stats[r][0], inv = stats[r][1];
        for (int c = p * 128; c < p * 128 + 128; ++c)
            feats[r][c] = f2bf((row[c] - mu) * inv * g_in[c] + bi_in[c]);
    }
    __syncthreads();
    // --- LayerNorm(pred) -> feats[:, 1024:2048]
    {
        int r = tid >> 3, p = tid & 7;
        const float* row = pred + (size_t)(m0 + r) * D;
        float s = 0.f, ss = 0.f;
        for (int c = p * 128; c < p * 128 + 128; ++c) { float x = row[c]; s += x; ss += x * x; }
        red[r][p][0] = s; red[r][p][1] = ss;
    }
    __syncthreads();
    if (tid < 16) {
        float s = 0.f, ss = 0.f;
        for (int p = 0; p < 8; ++p) { s += red[tid][p][0]; ss += red[tid][p][1]; }
        float mu = s * (1.f / D), var = ss * (1.f / D) - mu * mu;
        stats[tid][0] = mu; stats[tid][1] = rsqrtf(var + 1e-5f);
    }
    __syncthreads();
    {
        int r = tid >> 3, p = tid & 7;
        const float* row = pred + (size_t)(m0 + r) * D;
        float mu = stats[r][0], inv = stats[r][1];
        for (int c = p * 128; c < p * 128 + 128; ++c)
            feats[r][1024 + c] = f2bf((row[c] - mu) * inv * g_pr[c] + bi_pr[c]);
    }
    __syncthreads();

    // --- hmid = silu(feats @ W1[0:2048] + conf * W1[2048] + b1), via bf16 WMMA
    {
        const unsigned int* arow = (const unsigned int*)&feats[ln][0];   // ldk=1024 dw
        for (int g = 0; g < 4; ++g) {
            int nt0 = wv * 16 + g * 4;
            const unsigned int* b0c = w1T + (size_t)((nt0 + 0) * 16 + ln) * 1024;
            const unsigned int* b1c = w1T + (size_t)((nt0 + 1) * 16 + ln) * 1024;
            const unsigned int* b2c = w1T + (size_t)((nt0 + 2) * 16 + ln) * 1024;
            const unsigned int* b3c = w1T + (size_t)((nt0 + 3) * 16 + ln) * 1024;
            v8f c0 = {}, c1 = {}, c2 = {}, c3 = {};
            mma_n4<64>(arow, b0c, b1c, b2c, b3c, hf, c0, c1, c2, c3);
#pragma unroll
            for (int i = 0; i < 4; ++i) {
                int n = (nt0 + i) * 16 + ln;
                float w1c = W1f[(size_t)2048 * D + n];   // conf row of W1 (rank-1)
                float bb = b1[n];
                v8f c = (i == 0) ? c0 : (i == 1) ? c1 : (i == 2) ? c2 : c3;
#pragma unroll
                for (int v = 0; v < 8; ++v) {
                    int m = hf * 8 + v;
                    float x = c[v] + bb + confl[m] * w1c;
                    hmid[m][n] = x / (1.f + expf(-x));   // SiLU
                }
            }
        }
    }
    __syncthreads();
    // --- alpha = clip(sigmoid(hmid . W2 + b2))
    {
        int r = tid >> 3, p = tid & 7;
        float s = 0.f;
        for (int c = p * 128; c < p * 128 + 128; ++c) s += hmid[r][c] * W2[c];
        dotred[r][p] = s;
    }
    __syncthreads();
    if (tid < 16) {
        float s = 0.f;
        for (int p = 0; p < 8; ++p) s += dotred[tid][p];
        s += b2[0];
        float a = 1.f / (1.f + expf(-s));
        a = fminf(fmaxf(a, 0.f), 1.f);
        alphal[tid] = a;
        atomicAdd(mean_out, a * (1.f / (float)BL));
    }
    __syncthreads();
    // --- y = h + alpha * pred
    {
        int r = tid >> 3, p = tid & 7;
        float a = alphal[r];
        const float* hr = h + (size_t)(m0 + r) * D;
        const float* pr = pred + (size_t)(m0 + r) * D;
        float* yr = yout + (size_t)(m0 + r) * D;
        for (int c = p * 128; c < p * 128 + 128; ++c) yr[c] = hr[c] + a * pr[c];
    }
}

extern "C" void kernel_launch(void* const* d_in, const int* in_sizes, int n_in,
                              void* d_out, int out_size, void* d_ws, size_t ws_size,
                              hipStream_t stream) {
    const float* h_in      = (const float*)d_in[0];
    const float* Wk        = (const float*)d_in[1];
    const float* bk        = (const float*)d_in[2];
    const float* cube_keys = (const float*)d_in[3];
    const float* cube_vals = (const float*)d_in[4];
    const float* ln_in_g   = (const float*)d_in[5];
    const float* ln_in_b   = (const float*)d_in[6];
    const float* ln_pred_g = (const float*)d_in[7];
    const float* ln_pred_b = (const float*)d_in[8];
    const float* W1        = (const float*)d_in[9];
    const float* b1        = (const float*)d_in[10];
    const float* W2        = (const float*)d_in[11];
    const float* b2        = (const float*)d_in[12];
    float* out = (float*)d_out;

    char* ws = (char*)d_ws;
    size_t o = 0;
    unsigned int* mn  = (unsigned int*)(ws + o); o += (size_t)NSLOTS * D * 2;  // 64 MB
    unsigned int* kn  = (unsigned int*)(ws + o); o += (size_t)BL * D * 2;      // 16 MB
    unsigned int* hb  = (unsigned int*)(ws + o); o += (size_t)BL * D * 2;      // 16 MB
    unsigned int* wkT = (unsigned int*)(ws + o); o += (size_t)D * D * 2;       //  2 MB
    unsigned int* w1T = (unsigned int*)(ws + o); o += (size_t)D * 2048 * 2;    //  4 MB
    float* pred  = (float*)(ws + o); o += (size_t)BL * D * 4;                  // 32 MB
    float* confb = (float*)(ws + o); o += (size_t)BL * 4;
    float* mean_out = out + (size_t)BL * D;   // [mean_alpha, mean_conf]

    hipLaunchKernelGGL(cgb_init, dim3(1), dim3(32), 0, stream, mean_out);
    hipLaunchKernelGGL(cgb_cvt_h,   dim3((BL * D / 2) / 256), dim3(256), 0, stream, h_in, hb);
    hipLaunchKernelGGL(cgb_cvt_wkT, dim3((D * (D / 2)) / 256), dim3(256), 0, stream, Wk, wkT);
    hipLaunchKernelGGL(cgb_cvt_w1T, dim3((D * 1024) / 256), dim3(256), 0, stream, W1, w1T);
    hipLaunchKernelGGL(cgb_norm_cube, dim3(NSLOTS), dim3(256), 0, stream, cube_keys, mn);
    hipLaunchKernelGGL(cgb_proj, dim3(NBLK), dim3(128), 0, stream, hb, wkT, bk, kn);
    hipLaunchKernelGGL(cgb_sims_topk, dim3(NBLK32), dim3(128), 0, stream,
                       kn, mn, cube_vals, pred, confb, mean_out);
    hipLaunchKernelGGL(cgb_gate, dim3(NBLK), dim3(128), 0, stream,
                       h_in, pred, confb, w1T, W1, b1, W2, b2,
                       ln_in_g, ln_in_b, ln_pred_g, ln_pred_b, out, mean_out);
}
